// BiLSTM_46720654246470
// MI455X (gfx1250) — matrix operand order repaired and checked by reading, not compiled, not run on hardware
//
#include <hip/hip_runtime.h>
#include <cstdint>

// ---------------- CDNA5 WMMA types ----------------
typedef __bf16 v16bf __attribute__((ext_vector_type(16)));
typedef float  v8f   __attribute__((ext_vector_type(8)));

union Frag {
    unsigned int u[8];
    v16bf        v;
};

__device__ __forceinline__ unsigned int f2bf(float f) {
    unsigned int x = __float_as_uint(f);
    return (x + 0x7FFFu + ((x >> 16) & 1u)) >> 16;   // RNE to bf16
}
__device__ __forceinline__ unsigned int pack2(float lo, float hi) {
    return f2bf(lo) | (f2bf(hi) << 16);
}
__device__ __forceinline__ float bf2f(unsigned short u) {
    return __uint_as_float(((unsigned int)u) << 16);
}
__device__ __forceinline__ float sigf(float x) {
    return 1.0f / (1.0f + __expf(-x));
}

// Problem constants
#define NB   32      // batch
#define SS   256     // sequence
#define DD   1024    // input dim
#define HH   1024    // hidden dim

// B matrices are stored fragment-major: for WMMA tile (kt: 32 K-values,
// ct: 16 cols), lane L holds 8 consecutive uint32 (bf16 pairs) in exactly
// contiguous k order (k = 16*(L>=16) + elem):  pair q(v) = kt*16 + (L>=16)*8 + v,
// col = ct*16 + (L&15).  -> each fragment = 2x global_load_b128.

// ---------------- Phase 0: convert / repack / zero-state ----------------
__global__ __launch_bounds__(256) void bilstm_convert(
    const float* __restrict__ x, const float* __restrict__ wih,
    const float* __restrict__ whh,
    uint32_t* __restrict__ xpack, uint32_t* __restrict__ wihp,
    uint32_t* __restrict__ whhp, float* __restrict__ hbuf,
    float* __restrict__ cbuf)
{
    const long long i = (long long)blockIdx.x * 256 + threadIdx.x;
    const long long NXP = (long long)NB * SS * (DD / 2);          // 4,194,304
    const long long NWI = (long long)(DD / 2) * 8192;             // 4,194,304
    const long long NWH = 2LL * (HH / 2) * 4096;                  // 4,194,304
    if (i < NXP) {
        // x: row-major bf16 pairs along d  (A operand, 2x b128 per frag)
        long long row = i >> 9;            // / 512
        int d2 = (int)(i & 511);
        const float* p = x + row * DD + 2 * d2;
        xpack[i] = pack2(p[0], p[1]);
    } else if (i < NXP + NWI) {
        // W_ih: fragment-major tiles, K=1024 (KT=32), C=8192 (CT=512)
        long long e = i - NXP;
        int v    = (int)(e & 7);
        int lane = (int)((e >> 3) & 31);
        long long tidx = e >> 8;
        int ct = (int)(tidx & 511);
        int kt = (int)(tidx >> 9);
        int q    = kt * 16 + (lane >> 4) * 8 + v;   // bf16-pair row (d/2)
        int colG = ct * 16 + (lane & 15);
        int g = colG >> 10, h = colG & 1023;
        const float* p = wih + ((long long)(g * DD + 2 * q)) * HH + h;
        wihp[e] = pack2(p[0], p[HH]);
    } else if (i < NXP + NWI + NWH) {
        // W_hh: fragment-major tiles per dir, K=1024 (KT=32), C=4096 (CT=256)
        long long e = i - NXP - NWI;
        int v    = (int)(e & 7);
        int lane = (int)((e >> 3) & 31);
        long long tidx = e >> 8;
        int ct  = (int)(tidx & 255);
        int kt  = (int)((tidx >> 8) & 31);
        int dir = (int)(tidx >> 13);
        int q   = kt * 16 + (lane >> 4) * 8 + v;
        int col = ct * 16 + (lane & 15);
        int g = col >> 10, h = col & 1023;
        int gp = dir * 4 + g;
        const float* p = whh + ((long long)(gp * HH + 2 * q)) * HH + h;
        whhp[e] = pack2(p[0], p[HH]);
    } else {
        long long z = i - NXP - NWI - NWH;
        if (z < 131072) hbuf[z] = 0.0f;            // both h slots
        else            cbuf[z - 131072] = 0.0f;   // c state
    }
}

// ---------------- Phase 1: x_h = x @ W_ih  (bf16 WMMA, f32 accum) --------
// Block tile 64x256, 8 waves; each wave: 2 m-tiles x 4 n-tiles = 8 WMMA/K-step.
// Output xh (bf16): index ((s*8 + g)*32 + n)*1024 + h
__global__ __launch_bounds__(256) void bilstm_xgemm(
    const uint32_t* __restrict__ xpack, const uint32_t* __restrict__ wihp,
    unsigned short* __restrict__ xh)
{
    const int lane = threadIdx.x & 31;
    const int w    = threadIdx.x >> 5;
    const int half = lane >> 4;
    const int l15  = lane & 15;
    const int mp   = w & 1;                 // wave's m-pair (tiles 2mp,2mp+1)
    const int ng   = w >> 1;                // wave's n-group (4 tiles)
    const int m0   = blockIdx.x * 64;
    const int n0   = blockIdx.y * 256;
    const int ct0  = (n0 >> 4) + ng * 4;

    v8f acc[2][4] = {};

    for (int kb = 0; kb < DD; kb += 32) {
        const int kt = kb >> 5;
        const int kp = kb >> 1;
        Frag a[2];
#pragma unroll
        for (int mt = 0; mt < 2; ++mt) {
            const int row = m0 + (mp * 2 + mt) * 16 + l15;
            const uint32_t* ap = xpack + (long long)row * (DD / 2) + kp + half * 8;
            *(uint4*)&a[mt].u[0] = *(const uint4*)ap;
            *(uint4*)&a[mt].u[4] = *(const uint4*)(ap + 4);
        }
#pragma unroll
        for (int nt = 0; nt < 4; ++nt) {
            Frag b;
            const uint32_t* bp =
                wihp + (((long long)(kt * 512 + ct0 + nt)) * 32 + lane) * 8;
            *(uint4*)&b.u[0] = ((const uint4*)bp)[0];
            *(uint4*)&b.u[4] = ((const uint4*)bp)[1];
            acc[0][nt] = __builtin_amdgcn_wmma_f32_16x16x32_bf16(
                false, a[0].v, false, b.v, (short)0, acc[0][nt], false, false);
            acc[1][nt] = __builtin_amdgcn_wmma_f32_16x16x32_bf16(
                false, a[1].v, false, b.v, (short)0, acc[1][nt], false, false);
        }
    }
#pragma unroll
    for (int mt = 0; mt < 2; ++mt) {
#pragma unroll
        for (int nt = 0; nt < 4; ++nt) {
            const int colG = n0 + (ng * 4 + nt) * 16 + l15;
            const int g = colG >> 10, h = colG & 1023;
#pragma unroll
            for (int r = 0; r < 8; ++r) {
                const int M = m0 + (mp * 2 + mt) * 16 + r + half * 8;
                const int n = M >> 8, s = M & 255;   // row = n*S + s
                xh[((long long)(s * 8 + g) * 32 + n) * 1024 + h] =
                    (unsigned short)f2bf(acc[mt][nt][r]);
            }
        }
    }
}

// ---------------- Phase 2: one recurrence step (both directions) ---------
// 64 blocks: dir = bx>>5, column tile jt = bx&31 (32 hidden cols each).
// GEMM: M=32(batch) x N=128(4 gates x 32 cols) x K=1024(h_prev), then fused
// LSTM nonlinearity + state update + output stores.  h double-buffered.
__global__ __launch_bounds__(256) void bilstm_step(
    const uint32_t* __restrict__ whhp, const unsigned short* __restrict__ xh,
    const float* __restrict__ bias, const float* __restrict__ mask,
    float* __restrict__ hbuf, float* __restrict__ cbuf,
    float* __restrict__ out, int t)
{
    __shared__ float pre[4 * 32 * 32];      // [gate][n][jrel], 16 KB

    const int lane = threadIdx.x & 31;
    const int w    = threadIdx.x >> 5;
    const int half = lane >> 4;
    const int l15  = lane & 15;
    const int dir  = blockIdx.x >> 5;
    const int j0   = (blockIdx.x & 31) * 32;
    const int g    = w >> 1;                // gate handled by this wave
    const int mi   = w & 1;                 // m-tile (batch rows 16*mi..)

    const float*    hread  = hbuf + (t & 1) * 65536 + dir * 32768;
    float*          hwrite = hbuf + ((t + 1) & 1) * 65536 + dir * 32768;
    const uint32_t* wp     = whhp + (long long)dir * 2097152;  // 32*256*256
    const int ct0 = g * 64 + (j0 >> 4);

    v8f acc[2] = {};
    for (int kb = 0; kb < HH; kb += 32) {
        Frag a;
        const float* hp = hread + (mi * 16 + l15) * HH + kb + half * 16;
        float4 f0 = *(const float4*)(hp);
        float4 f1 = *(const float4*)(hp + 4);
        float4 f2 = *(const float4*)(hp + 8);
        float4 f3 = *(const float4*)(hp + 12);
        a.u[0] = pack2(f0.x, f0.y); a.u[1] = pack2(f0.z, f0.w);
        a.u[2] = pack2(f1.x, f1.y); a.u[3] = pack2(f1.z, f1.w);
        a.u[4] = pack2(f2.x, f2.y); a.u[5] = pack2(f2.z, f2.w);
        a.u[6] = pack2(f3.x, f3.y); a.u[7] = pack2(f3.z, f3.w);
        const int kt = kb >> 5;
#pragma unroll
        for (int nt = 0; nt < 2; ++nt) {
            Frag b;
            const uint32_t* bp =
                wp + (((long long)(kt * 256 + ct0 + nt)) * 32 + lane) * 8;
            *(uint4*)&b.u[0] = ((const uint4*)bp)[0];
            *(uint4*)&b.u[4] = ((const uint4*)bp)[1];
            acc[nt] = __builtin_amdgcn_wmma_f32_16x16x32_bf16(
                false, a.v, false, b.v, (short)0, acc[nt], false, false);
        }
    }
#pragma unroll
    for (int nt = 0; nt < 2; ++nt) {
#pragma unroll
        for (int r = 0; r < 8; ++r) {
            const int n = mi * 16 + r + half * 8;
            const int jr = nt * 16 + l15;
            pre[(g * 32 + n) * 32 + jr] = acc[nt][r];
        }
    }
    __syncthreads();

    const int s = (dir == 0) ? t : (SS - 1 - t);
    for (int e = threadIdx.x; e < 32 * 32; e += 256) {
        const int n = e >> 5, jr = e & 31;
        const int j = j0 + jr;
        const long long xb = ((long long)(s * 8 + dir * 4) * 32 + n) * 1024 + j;
        const float pf = pre[(0 * 32 + n) * 32 + jr] + bf2f(xh[xb])           + bias[(dir * 4 + 0) * HH + j];
        const float pi = pre[(1 * 32 + n) * 32 + jr] + bf2f(xh[xb + 32768])   + bias[(dir * 4 + 1) * HH + j];
        const float po = pre[(2 * 32 + n) * 32 + jr] + bf2f(xh[xb + 65536])   + bias[(dir * 4 + 2) * HH + j];
        const float pg = pre[(3 * 32 + n) * 32 + jr] + bf2f(xh[xb + 98304])   + bias[(dir * 4 + 3) * HH + j];

        const float fg = sigf(pf), ig = sigf(pi), og = sigf(po);
        const long long ci = (long long)dir * 32768 + n * HH + j;
        const float cnew = fg * cbuf[ci] + ig * tanhf(pg);
        const float hnew = og * tanhf(cnew);
        const float mt   = mask[n * SS + s];
        const float hold = hread[n * HH + j];
        const float hid  = (mt == 0.0f) ? hold : hnew;

        cbuf[ci] = cnew;
        hwrite[n * HH + j] = hid;
        out[((long long)n * SS + s) * 2048 + dir * 1024 + j] = hid * mt;
        if (t == SS - 1)
            out[(long long)NB * SS * 2048 + (long long)n * 2048 + dir * 1024 + j] = hid;
    }
}

// ---------------- Host launch ----------------
extern "C" void kernel_launch(void* const* d_in, const int* in_sizes, int n_in,
                              void* d_out, int out_size, void* d_ws, size_t ws_size,
                              hipStream_t stream) {
    const float* x    = (const float*)d_in[0];
    const float* mask = (const float*)d_in[1];
    const float* wih  = (const float*)d_in[2];
    const float* whh  = (const float*)d_in[3];
    const float* bias = (const float*)d_in[4];
    float* out = (float*)d_out;
    char*  ws  = (char*)d_ws;

    uint32_t*       xpack = (uint32_t*)(ws + 0);
    uint32_t*       wihp  = (uint32_t*)(ws + 16777216LL);
    uint32_t*       whhp  = (uint32_t*)(ws + 33554432LL);
    unsigned short* xh    = (unsigned short*)(ws + 50331648LL);
    float*          hbuf  = (float*)(ws + 184549376LL);   // 2 slots x 2 dirs x 32 x 1024
    float*          cbuf  = (float*)(ws + 185073664LL);   // 2 dirs x 32 x 1024

    // Phase 0: repack to WMMA fragment-major bf16 layouts + zero states
    bilstm_convert<<<49920, 256, 0, stream>>>(x, wih, whh, xpack, wihp, whhp,
                                              hbuf, cbuf);
    // Phase 1: x_h = x @ W_ih  (8192 x 8192 x 1024 bf16 WMMA GEMM)
    dim3 g1(128, 32);
    bilstm_xgemm<<<g1, 256, 0, stream>>>(xpack, wihp, xh);
    // Phase 2: 256 sequential recurrence steps (fwd + bwd fused per launch)
    for (int t = 0; t < SS; ++t)
        bilstm_step<<<64, 256, 0, stream>>>(whhp, xh, bias, mask, hbuf, cbuf,
                                            out, t);
}
